// DualAttention_57123065037080
// MI455X (gfx1250) — matrix operand order repaired and checked by reading, not compiled, not run on hardware
//
#include <hip/hip_runtime.h>

// ---------------------------------------------------------------------------
// Tacotron2-style decoder for MI455X (gfx1250, wave32, WMMA).
// Step-invariant weights packed once to bf16 WMMA tiles; the 400 sequential
// steps run as (packA -> fused WMMA GEMM+LSTM) x2 + attention + projection.
// GEMM K-loops are compile-time unrolled (immediate-offset loads, no per-iter
// address math) and register double-buffered so WMMAs overlap tile loads.
// ---------------------------------------------------------------------------

typedef __attribute__((ext_vector_type(16))) __bf16 v16bf;
typedef __attribute__((ext_vector_type(8)))  float  v8f;

#define WMMA_BF16(a, b, c) \
  __builtin_amdgcn_wmma_f32_16x16x32_bf16(false, (a), false, (b), (short)0, (c), false, false)

// fp32 -> bf16, round-to-nearest-even
__device__ __forceinline__ unsigned short f2bf(float x) {
  unsigned int u = __float_as_uint(x);
  u += 0x7fffu + ((u >> 16) & 1u);
  return (unsigned short)(u >> 16);
}

// A-operand (16x32 bf16, MxK) element mapping: lane,e -> k   (m = lane & 15)
// ISA 7.12.2: lanes0-15 K=0-7/16-23, lanes16-31 K=8-15/24-31, pairs per VGPR.
__device__ __forceinline__ int a_kmap(int lane, int e) {
  return ((lane >> 4) << 3) | ((e >> 3) << 4) | (((e >> 1) & 3) << 1) | (e & 1);
}

// C/D-matrix (16x16 f32) packed state: tile = mt*ntiles+nt, per-lane v8f.
// element (m,n): lane = (n&15) + 16*(m&8 ? 1:0), r = m&7
__device__ __forceinline__ size_t idxC(int m, int n, int ntiles) {
  int mt = m >> 4, mr = m & 15, nt = n >> 4, nl = n & 15;
  return ((size_t)(mt * ntiles + nt) << 8) + (size_t)((nl + ((mr >> 3) << 4)) << 3) + (mr & 7);
}

// ---------------------------------------------------------------------------
// Weight pack: rows n of [W1 | W2] -> B-operand tiles [nt][kt][lane][e] (n = lane&15, k = a_kmap(lane,e)).
// k < K1: col = k (+1 if k>=skip_at, to skip attn_wih's always-zero column).
// ---------------------------------------------------------------------------
__global__ void __launch_bounds__(256) k_packB(unsigned short* __restrict__ dst,
    const float* __restrict__ W1, int ld1, int K1, int skip_at,
    const float* __restrict__ W2, int ld2, int K2, int KT, int NT) {
  size_t tid = (size_t)blockIdx.x * 256 + threadIdx.x;
  int e = (int)(tid & 15), lane = (int)((tid >> 4) & 31);
  size_t tile = tid >> 9;
  int kt = (int)(tile % KT), nt = (int)(tile / KT);
  if (nt >= NT) return;
  int k = kt * 32 + a_kmap(lane, e); // B layout: k order matches A (a_kmap)
  int n = nt * 16 + (lane & 15);     // lane & 15 == n
  float v = 0.f;
  if (k < K1)            v = W1[(size_t)n * ld1 + k + (k >= skip_at ? 1 : 0)];
  else if (k < K1 + K2)  v = W2[(size_t)n * ld2 + (k - K1)];
  dst[tid] = f2bf(v);
}

// Pack row-major f32 [M x K] into A-operand tiles (zero-pad OOB k).
__global__ void __launch_bounds__(256) k_packA_rm(unsigned short* __restrict__ dst,
    const float* __restrict__ src, int M, int K, int KT) {
  size_t tid = (size_t)blockIdx.x * 256 + threadIdx.x;
  int e = (int)(tid & 15), lane = (int)((tid >> 4) & 31);
  size_t tile = tid >> 9;
  int kt = (int)(tile % KT), mt = (int)(tile / KT);
  int m = mt * 16 + (lane & 15);
  int k = kt * 32 + a_kmap(lane, e);
  float v = (m < M && k < K) ? src[(size_t)m * K + k] : 0.f;
  dst[tid] = f2bf(v);
}

// Pack teacher-forced targets: row m = t*32+b, 80 mel cols (pad to 96). t==0 -> 0.
__global__ void __launch_bounds__(256) k_packA_tgt(unsigned short* __restrict__ dst,
    const float* __restrict__ targets) {
  size_t tid = (size_t)blockIdx.x * 256 + threadIdx.x;
  int e = (int)(tid & 15), lane = (int)((tid >> 4) & 31);
  size_t tile = tid >> 9;
  int kt = (int)(tile % 3), mt = (int)(tile / 3);
  int m = mt * 16 + (lane & 15);
  int t = m >> 5, b = m & 31;
  int k = kt * 32 + a_kmap(lane, e);
  float v = 0.f;
  if (k < 80 && t > 0) v = targets[((size_t)b * 80 + k) * 400 + (t - 1)];
  dst[tid] = f2bf(v);
}

// Per-step A for attn LSTM: [frame(256) | ctx(768) | ah(1024)], KT=64.
__global__ void __launch_bounds__(256) k_packA_attn(unsigned short* __restrict__ dst,
    const float* __restrict__ frames, int step,
    const float* __restrict__ ctx, const float* __restrict__ ahC) {
  size_t tid = (size_t)blockIdx.x * 256 + threadIdx.x;
  int e = (int)(tid & 15), lane = (int)((tid >> 4) & 31);
  int tile = (int)(tid >> 9);
  int kt = tile % 64, mt = tile / 64;
  int b = mt * 16 + (lane & 15);
  int k = kt * 32 + a_kmap(lane, e);
  float v;
  if (k < 256)        v = frames[((size_t)(step * 32 + b)) * 256 + k];
  else if (k < 1024)  v = ctx[(size_t)b * 768 + (k - 256)];
  else                v = ahC[idxC(b, k - 1024, 64)];
  dst[tid] = f2bf(v);
}

// Per-step A for dec LSTM: [ah(1024) | ctx(768) | dh(1024)], KT=88.
__global__ void __launch_bounds__(256) k_packA_dec(unsigned short* __restrict__ dst,
    const float* __restrict__ ahC, const float* __restrict__ ctx,
    const float* __restrict__ dhC) {
  size_t tid = (size_t)blockIdx.x * 256 + threadIdx.x;
  int e = (int)(tid & 15), lane = (int)((tid >> 4) & 31);
  int tile = (int)(tid >> 9);
  int kt = tile % 88, mt = tile / 88;
  int b = mt * 16 + (lane & 15);
  int k = kt * 32 + a_kmap(lane, e);
  float v;
  if (k < 1024)       v = ahC[idxC(b, k, 64)];
  else if (k < 1792)  v = ctx[(size_t)b * 768 + (k - 1024)];
  else                v = dhC[idxC(b, k - 1792, 64)];
  dst[tid] = f2bf(v);
}

// ---------------------------------------------------------------------------
// Fused LSTM: each wave owns one (mt, c-column tile); accumulates the 4 gate
// tiles (i,f,g,o live at n, n+1024, n+2048, n+3072), then does the cell update
// in the native C-layout registers. KT is compile-time: the K loop fully
// unrolls so every tile load is a constant-IOFFSET global_load_b128 (max
// offset KT*1KB <= 90KB << 24-bit IOFFSET); the double-buffer rotation becomes
// register renaming, guaranteeing >=1 tile of load lookahead per WMMA group.
// grid 16 x 256 (2 mt * 64 c-tiles).
// ---------------------------------------------------------------------------
template <int KT>
__global__ void __launch_bounds__(256) k_gemm_lstm(
    const unsigned short* __restrict__ Ap, const unsigned short* __restrict__ Bp,
    const float* __restrict__ bih, const float* __restrict__ bhh,
    float* __restrict__ cC, float* __restrict__ hC) {
  const int lane = threadIdx.x & 31;
  const int wave = (blockIdx.x << 3) + (threadIdx.x >> 5);
  const int mt = wave >> 6;
  const int nc = wave & 63;
  const v16bf* Av = (const v16bf*)Ap + (size_t)mt * KT * 32 + lane;
  const v16bf* B0 = (const v16bf*)Bp + ((size_t)(nc)       * KT) * 32 + lane;
  const v16bf* B1 = (const v16bf*)Bp + ((size_t)(64 + nc)  * KT) * 32 + lane;
  const v16bf* B2 = (const v16bf*)Bp + ((size_t)(128 + nc) * KT) * 32 + lane;
  const v16bf* B3 = (const v16bf*)Bp + ((size_t)(192 + nc) * KT) * 32 + lane;
  // one-shot warm of the head of each weight stream (L2 -> WGP$)
  __builtin_prefetch((const void*)B0, 0, 1);
  __builtin_prefetch((const void*)B1, 0, 1);
  __builtin_prefetch((const void*)B2, 0, 1);
  __builtin_prefetch((const void*)B3, 0, 1);
  v8f ai = {}, af2 = {}, ag = {}, ao = {};
  v16bf a  = Av[0];
  v16bf b0 = B0[0], b1 = B1[0], b2 = B2[0], b3 = B3[0];
#pragma unroll
  for (int kt = 0; kt < KT - 1; ++kt) {
    const int nx = (kt + 1) * 32;      // constant after unroll -> imm offsets
    v16bf an = Av[nx];
    v16bf c0 = B0[nx], c1 = B1[nx], c2 = B2[nx], c3 = B3[nx];
    ai  = WMMA_BF16(a, b0, ai);
    af2 = WMMA_BF16(a, b1, af2);
    ag  = WMMA_BF16(a, b2, ag);
    ao  = WMMA_BF16(a, b3, ao);
    a = an; b0 = c0; b1 = c1; b2 = c2; b3 = c3;
  }
  ai  = WMMA_BF16(a, b0, ai);
  af2 = WMMA_BF16(a, b1, af2);
  ag  = WMMA_BF16(a, b2, ag);
  ao  = WMMA_BF16(a, b3, ao);

  const int nb = nc * 16 + (lane & 15);
  const float bi = bih[nb]        + bhh[nb];
  const float bf = bih[1024 + nb] + bhh[1024 + nb];
  const float bg = bih[2048 + nb] + bhh[2048 + nb];
  const float bo = bih[3072 + nb] + bhh[3072 + nb];
  const size_t base = ((size_t)(mt * 64 + nc) << 8) + (size_t)lane * 8;
  v8f cold = *(const v8f*)(cC + base);
  v8f cnew, hnew;
#pragma unroll
  for (int j = 0; j < 8; ++j) {
    float i = ai[j] + bi, f = af2[j] + bf, g = ag[j] + bg, o = ao[j] + bo;
    float si = 1.f / (1.f + __expf(-i));
    float sf = 1.f / (1.f + __expf(-f));
    float so = 1.f / (1.f + __expf(-o));
    float c = sf * cold[j] + si * tanhf(g);
    cnew[j] = c;
    hnew[j] = so * tanhf(c);
  }
  *(v8f*)(cC + base) = cnew;
  *(v8f*)(hC + base) = hnew;
}

// Generic packed-tile GEMM -> row-major f32 (optional relu), K fully unrolled.
template <int KT>
__global__ void __launch_bounds__(256) k_gemm_rm(
    const unsigned short* __restrict__ Ap, const unsigned short* __restrict__ Bp,
    float* __restrict__ out, int MT, int NT, int N, int relu) {
  const int lane = threadIdx.x & 31;
  const int w = (blockIdx.x << 3) + (threadIdx.x >> 5);
  if (w >= MT * NT) return;          // wave-uniform exit (grids launched exact)
  const int mt = w / NT, nt = w % NT;
  const v16bf* Av = (const v16bf*)Ap + (size_t)mt * KT * 32 + lane;
  const v16bf* Bv = (const v16bf*)Bp + (size_t)nt * KT * 32 + lane;
  __builtin_prefetch((const void*)Av, 0, 1);
  __builtin_prefetch((const void*)Bv, 0, 1);
  v8f acc = {};
  v16bf a = Av[0], b = Bv[0];
#pragma unroll
  for (int kt = 0; kt < KT - 1; ++kt) {
    const int nx = (kt + 1) * 32;
    v16bf an = Av[nx], bn = Bv[nx];
    acc = WMMA_BF16(a, b, acc);
    a = an; b = bn;
  }
  acc = WMMA_BF16(a, b, acc);
  const int n = nt * 16 + (lane & 15);
  const int mbase = mt * 16 + ((lane >> 4) << 3);
#pragma unroll
  for (int r = 0; r < 8; ++r) {
    float v = acc[r];
    if (relu) v = fmaxf(v, 0.f);
    out[(size_t)(mbase + r) * N + n] = v;
  }
}

// ---------------------------------------------------------------------------
// Attention step: location conv + tanh energies + masked softmax + context.
// One block per batch row; LDS for q, weights, reductions.
// ---------------------------------------------------------------------------
__global__ void __launch_bounds__(256) k_attention(
    const float* __restrict__ ahC, const float* __restrict__ mem,
    const float* __restrict__ memproc, const int* __restrict__ tlen,
    const float* __restrict__ loc_conv, const float* __restrict__ loc_lin,
    const float* __restrict__ wq, const float* __restrict__ vvec,
    float* __restrict__ aw, float* __restrict__ awc, float* __restrict__ ctx,
    float* __restrict__ align_out, int step) {
  const int b = blockIdx.x, tid = threadIdx.x;
  __shared__ float s_aw[192], s_awc[192], s_q[128], s_w[192], s_red[256];
  if (tid < 192) { s_aw[tid] = aw[b * 192 + tid]; s_awc[tid] = awc[b * 192 + tid]; }
  __syncthreads();
  if (tid < 128) {
    float acc = 0.f;
    const float* wr = wq + (size_t)tid * 1024;
    for (int j = 0; j < 1024; ++j) acc += ahC[idxC(b, j, 64)] * wr[j];
    s_q[tid] = acc;
  }
  __syncthreads();
  float e = -1e30f;
  if (tid < 192) {
    float lf[32];
#pragma unroll
    for (int f = 0; f < 32; ++f) lf[f] = 0.f;
    for (int kk = 0; kk < 31; ++kk) {
      int tt = tid + kk - 15;
      float a0 = (tt >= 0 && tt < 192) ? s_aw[tt]  : 0.f;
      float a1 = (tt >= 0 && tt < 192) ? s_awc[tt] : 0.f;
#pragma unroll
      for (int f = 0; f < 32; ++f)
        lf[f] += loc_conv[(f * 2 + 0) * 31 + kk] * a0 +
                 loc_conv[(f * 2 + 1) * 31 + kk] * a1;
    }
    float es = 0.f;
    const float* mp = memproc + ((size_t)b * 192 + tid) * 128;
    for (int a = 0; a < 128; ++a) {
      float la = 0.f;
      const float* lr = loc_lin + a * 32;
#pragma unroll
      for (int f = 0; f < 32; ++f) la += lr[f] * lf[f];
      es += vvec[a] * tanhf(s_q[a] + la + mp[a]);
    }
    e = (tid >= tlen[b]) ? -1e9f : es;
  }
  s_red[tid] = (tid < 192) ? e : -1e30f;
  __syncthreads();
  for (int s = 128; s > 0; s >>= 1) {
    if (tid < s) s_red[tid] = fmaxf(s_red[tid], s_red[tid + s]);
    __syncthreads();
  }
  float mx = s_red[0];
  __syncthreads();
  float p = (tid < 192) ? __expf(e - mx) : 0.f;
  s_red[tid] = p;
  __syncthreads();
  for (int s = 128; s > 0; s >>= 1) {
    if (tid < s) s_red[tid] += s_red[tid + s];
    __syncthreads();
  }
  float inv = 1.f / s_red[0];
  if (tid < 192) {
    float w = p * inv;
    s_w[tid] = w;
    aw[b * 192 + tid]  = w;
    awc[b * 192 + tid] = s_awc[tid] + w;
    align_out[((size_t)b * 400 + step) * 192 + tid] = w;
  }
  __syncthreads();
  for (int col = tid; col < 768; col += 256) {
    float acc = 0.f;
    const float* mrow = mem + (size_t)b * 192 * 768 + col;
    for (int t = 0; t < 192; ++t) acc += s_w[t] * mrow[(size_t)t * 768];
    ctx[(size_t)b * 768 + col] = acc;
  }
}

// Projection + gate: cat = [dh(1024) | ctx(768)]; 80 mel outs + 1 gate.
__global__ void __launch_bounds__(128) k_proj(
    const float* __restrict__ dhC, const float* __restrict__ ctx,
    const float* __restrict__ proj_w, const float* __restrict__ proj_b,
    const float* __restrict__ gate_w, const float* __restrict__ gate_b,
    float* __restrict__ out, int step) {
  const int b = blockIdx.x, tid = threadIdx.x;
  if (tid > 80) return;
  const float* wrow = (tid < 80) ? (proj_w + (size_t)tid * 1792) : gate_w;
  float acc = (tid < 80) ? proj_b[tid] : gate_b[0];
  for (int j = 0; j < 1024; ++j) acc += dhC[idxC(b, j, 64)] * wrow[j];
  const float* cx = ctx + (size_t)b * 768;
  for (int j = 0; j < 768; ++j) acc += cx[j] * wrow[1024 + j];
  if (tid < 80) out[((size_t)b * 80 + tid) * 400 + step] = acc;
  else          out[(size_t)32 * 80 * 400 + (size_t)b * 400 + step] = acc;
}

__global__ void k_zero(float* p, int n) {
  int i = blockIdx.x * 256 + threadIdx.x;
  if (i < n) p[i] = 0.f;
}

// ---------------------------------------------------------------------------
extern "C" void kernel_launch(void* const* d_in, const int* in_sizes, int n_in,
                              void* d_out, int out_size, void* d_ws, size_t ws_size,
                              hipStream_t stream) {
  (void)in_sizes; (void)n_in; (void)out_size; (void)ws_size;
  const float* text_emb  = (const float*)d_in[0];
  const int*   tlen      = (const int*)  d_in[1];
  const float* targets   = (const float*)d_in[2];
  const float* pn_w1     = (const float*)d_in[3];
  const float* pn_w2     = (const float*)d_in[4];
  const float* attn_wih  = (const float*)d_in[5];
  const float* attn_whh  = (const float*)d_in[6];
  const float* attn_bih  = (const float*)d_in[7];
  const float* attn_bhh  = (const float*)d_in[8];
  const float* wq        = (const float*)d_in[9];
  const float* wmem      = (const float*)d_in[10];
  const float* vvec      = (const float*)d_in[11];
  const float* loc_conv  = (const float*)d_in[12];
  const float* loc_lin   = (const float*)d_in[13];
  const float* dec_wih   = (const float*)d_in[14];
  const float* dec_whh   = (const float*)d_in[15];
  const float* dec_bih   = (const float*)d_in[16];
  const float* dec_bhh   = (const float*)d_in[17];
  const float* proj_w    = (const float*)d_in[18];
  const float* proj_b    = (const float*)d_in[19];
  const float* gate_w    = (const float*)d_in[20];
  const float* gate_b    = (const float*)d_in[21];
  float* out = (float*)d_out;
  float* align_out = out + (size_t)32 * 80 * 400 + (size_t)32 * 400;

  char* ws = (char*)d_ws;
  size_t cur = 0;
  auto alloc = [&](size_t bytes) -> char* {
    char* p = ws + cur;
    cur += (bytes + 255) & ~(size_t)255;
    return p;
  };
  // --- persistent small state (allocated first, contiguous, zeroed each call)
  float* AH  = (float*)alloc(32 * 1024 * 4);   // attn h, C-packed
  float* AC  = (float*)alloc(32 * 1024 * 4);   // attn c
  float* DH  = (float*)alloc(32 * 1024 * 4);   // dec h
  float* DC  = (float*)alloc(32 * 1024 * 4);   // dec c
  float* CTX = (float*)alloc(32 * 768 * 4);    // context, row-major
  float* AW  = (float*)alloc(32 * 192 * 4);    // prev attention weights
  float* AWC = (float*)alloc(32 * 192 * 4);    // cumulative weights
  const int STATE_F32 = 4 * 32 * 1024 + 32 * 768 + 2 * 32 * 192;  // 167936
  // --- packed bf16 weights
  unsigned short* BW_ATTN = (unsigned short*)alloc((size_t)256 * 64 * 512 * 2);
  unsigned short* BW_DEC  = (unsigned short*)alloc((size_t)256 * 88 * 512 * 2);
  unsigned short* BW_MEM  = (unsigned short*)alloc((size_t)8 * 24 * 512 * 2);
  unsigned short* BW_PN1  = (unsigned short*)alloc((size_t)16 * 3 * 512 * 2);
  unsigned short* BW_PN2  = (unsigned short*)alloc((size_t)16 * 8 * 512 * 2);
  // --- precompute buffers
  unsigned short* A_TGT  = (unsigned short*)alloc((size_t)800 * 3 * 512 * 2);
  float*          PRE1   = (float*)alloc((size_t)12800 * 256 * 4);
  unsigned short* A_PRE1 = (unsigned short*)alloc((size_t)800 * 8 * 512 * 2);
  float*          FRAMES = (float*)alloc((size_t)12800 * 256 * 4);
  unsigned short* A_MEM  = (unsigned short*)alloc((size_t)384 * 24 * 512 * 2);
  float*          MEMPROC = (float*)alloc((size_t)6144 * 128 * 4);
  // --- per-step A operands
  unsigned short* A_ATTN = (unsigned short*)alloc((size_t)2 * 64 * 512 * 2);
  unsigned short* A_DEC  = (unsigned short*)alloc((size_t)2 * 88 * 512 * 2);

  const int BIG = 1 << 30;

  // init recurrent state to zero (ws is poisoned by harness)
  k_zero<<<dim3((STATE_F32 + 255) / 256), dim3(256), 0, stream>>>(AH, STATE_F32);

  // pack weights: attn K=2048 (skip wih col 256 which always sees x==0)
  k_packB<<<dim3(256 * 64 * 2), dim3(256), 0, stream>>>(
      BW_ATTN, attn_wih, 1025, 1024, 256, attn_whh, 1024, 1024, 64, 256);
  k_packB<<<dim3(256 * 88 * 2), dim3(256), 0, stream>>>(
      BW_DEC, dec_wih, 1792, 1792, BIG, dec_whh, 1024, 1024, 88, 256);
  k_packB<<<dim3(8 * 24 * 2), dim3(256), 0, stream>>>(
      BW_MEM, wmem, 768, 768, BIG, nullptr, 1, 0, 24, 8);
  k_packB<<<dim3(16 * 3 * 2), dim3(256), 0, stream>>>(
      BW_PN1, pn_w1, 80, 80, BIG, nullptr, 1, 0, 3, 16);
  k_packB<<<dim3(16 * 8 * 2), dim3(256), 0, stream>>>(
      BW_PN2, pn_w2, 256, 256, BIG, nullptr, 1, 0, 8, 16);

  // prenet for all 400 frames (teacher forcing): 12800x80 -> 256 -> 256
  k_packA_tgt<<<dim3(800 * 3 * 2), dim3(256), 0, stream>>>(A_TGT, targets);
  k_gemm_rm<3><<<dim3(1600), dim3(256), 0, stream>>>(A_TGT, BW_PN1, PRE1, 800, 16, 256, 1);
  k_packA_rm<<<dim3(800 * 8 * 2), dim3(256), 0, stream>>>(A_PRE1, PRE1, 12800, 256, 8);
  k_gemm_rm<8><<<dim3(1600), dim3(256), 0, stream>>>(A_PRE1, BW_PN2, FRAMES, 800, 16, 256, 1);

  // mem_proc = text_embedding @ wmem.T  (6144x768 @ 768x128)
  k_packA_rm<<<dim3(384 * 24 * 2), dim3(256), 0, stream>>>(A_MEM, text_emb, 6144, 768, 24);
  k_gemm_rm<24><<<dim3(384), dim3(256), 0, stream>>>(A_MEM, BW_MEM, MEMPROC, 384, 8, 128, 0);

  // 400 sequential decoder steps
  for (int t = 0; t < 400; ++t) {
    k_packA_attn<<<dim3(2 * 64 * 2), dim3(256), 0, stream>>>(A_ATTN, FRAMES, t, CTX, AH);
    k_gemm_lstm<64><<<dim3(16), dim3(256), 0, stream>>>(
        A_ATTN, BW_ATTN, attn_bih, attn_bhh, AC, AH);
    k_attention<<<dim3(32), dim3(256), 0, stream>>>(
        AH, text_emb, MEMPROC, tlen, loc_conv, loc_lin, wq, vvec,
        AW, AWC, CTX, align_out, t);
    k_packA_dec<<<dim3(2 * 88 * 2), dim3(256), 0, stream>>>(A_DEC, AH, CTX, DH);
    k_gemm_lstm<88><<<dim3(16), dim3(256), 0, stream>>>(
        A_DEC, BW_DEC, dec_bih, dec_bhh, DC, DH);
    k_proj<<<dim3(32), dim3(128), 0, stream>>>(
        DH, CTX, proj_w, proj_b, gate_w, gate_b, out, t);
  }
}
